// PairWiseLearning_BARLOW_TWINS_65532611002848
// MI455X (gfx1250) — matrix operand order repaired and checked by reading, not compiled, not run on hardware
//
#include <hip/hip_runtime.h>
#include <hip/hip_bf16.h>

typedef float v2f __attribute__((ext_vector_type(2)));
typedef float v4f __attribute__((ext_vector_type(4)));
typedef float v8f __attribute__((ext_vector_type(8)));
typedef _Float16 v16h __attribute__((ext_vector_type(16)));

#define D_MODEL 128
#define WB_STRIDE 144   // float2 per kp-row; 144*2 dwords = 288 ≡ 32 (mod 64 banks)

// ---------------------------------------------------------------------------
// Degree / CSR construction
// ---------------------------------------------------------------------------
__global__ void k_zero2(int* __restrict__ cnt, int* __restrict__ cursor, int n) {
    int i = blockIdx.x * blockDim.x + threadIdx.x;
    if (i < n) { cnt[i] = 0; cursor[i] = 0; }
}

__global__ void k_count(const int* __restrict__ dst, int* __restrict__ cnt, int E) {
    int e = blockIdx.x * blockDim.x + threadIdx.x;
    if (e < E) atomicAdd(&cnt[dst[e]], 1);
}

#define SCAN_T 1024
__global__ __launch_bounds__(SCAN_T) void k_scan(const int* __restrict__ cnt,
                                                 int* __restrict__ row_ptr, int n) {
    __shared__ int partial[SCAN_T];
    int tid   = threadIdx.x;
    int chunk = (n + SCAN_T - 1) / SCAN_T;
    int begin = tid * chunk;
    int end   = begin + chunk; if (end > n) end = n;
    int s = 0;
    for (int i = begin; i < end; ++i) s += cnt[i];
    partial[tid] = s;
    __syncthreads();
    if (tid == 0) {
        int run = 0;
        for (int i = 0; i < SCAN_T; ++i) { int v = partial[i]; partial[i] = run; run += v; }
        row_ptr[n] = run;
    }
    __syncthreads();
    int run = partial[tid];
    for (int i = begin; i < end; ++i) { row_ptr[i] = run; run += cnt[i]; }
}

__global__ void k_dinv(const int* __restrict__ cnt, float* __restrict__ dinv, int n) {
    int i = blockIdx.x * blockDim.x + threadIdx.x;
    if (i < n) dinv[i] = rsqrtf((float)(cnt[i] + 1));   // +1 self loop, always > 0
}

__global__ void k_csr_fill(const int* __restrict__ src, const int* __restrict__ dst,
                           const int* __restrict__ row_ptr, int* __restrict__ cursor,
                           int* __restrict__ csr_src, int E) {
    int e = blockIdx.x * blockDim.x + threadIdx.x;
    if (e < E) {
        int d = dst[e];
        int p = atomicAdd(&cursor[d], 1);
        csr_src[row_ptr[d] + p] = src[e];
    }
}

// ---------------------------------------------------------------------------
// Embedding gather
// ---------------------------------------------------------------------------
__global__ void k_embed(const int* __restrict__ x, const float* __restrict__ emb,
                        float* __restrict__ feat, int n) {
    int idx = blockIdx.x * blockDim.x + threadIdx.x;
    if (idx < n * D_MODEL) {
        int node = idx >> 7, d = idx & 127;
        feat[idx] = emb[x[node] * D_MODEL + d];
    }
}

// ---------------------------------------------------------------------------
// WMMA GEMM: C[M,128] = A[M,128] @ W[128,128] (+ bias)
// W staged in LDS k-paired: one ds_load_b64 per B-fragment, bank-conflict-free.
// 8 waves/block, each wave owns a 16-row x 128-col strip (8 tiles of 16x16).
// ---------------------------------------------------------------------------
__global__ __launch_bounds__(256) void k_gemm128(const float* __restrict__ A,
                                                 const float* __restrict__ W,
                                                 const float* __restrict__ bias,
                                                 float* __restrict__ C, int M) {
    __shared__ v2f WB[64 * WB_STRIDE];   // WB[kp*STRIDE + n] = {W[2kp][n], W[2kp+1][n]}

    // cooperative staging of W (16384 floats)
    for (int i = threadIdx.x; i < D_MODEL * D_MODEL; i += 256) {
        int k = i >> 7, n = i & 127;
        ((float*)WB)[(((k >> 1) * WB_STRIDE) + n) * 2 + (k & 1)] = W[i];
    }
    __syncthreads();

    const int wave = threadIdx.x >> 5;
    const int lane = threadIdx.x & 31;
    const int m0   = (blockIdx.x * 8 + wave) * 16;
    if (m0 >= M) return;                       // wave-uniform exit (after barrier)
    const int lm = lane & 15;                  // 0..15
    const int lh = lane >> 4;                  // 0 or 1
    int arow = m0 + lm; if (arow >= M) arow = M - 1;   // clamp tail (stores guarded)
    const float* Ar = A + (size_t)arow * D_MODEL;

    v8f acc[8];
#pragma unroll
    for (int t = 0; t < 8; ++t) acc[t] = {};

#if __has_builtin(__builtin_amdgcn_wmma_f32_16x16x4_f32)
    // fp32 path: V_WMMA_F32_16X16X4_F32. A 16x4: M=lm, K pair = lh*2.
    for (int k0 = 0; k0 < D_MODEL; k0 += 4) {
        v2f a = *(const v2f*)(Ar + k0 + lh * 2);      // global, pipelined
        const v2f* wrow = &WB[(size_t)((k0 >> 1) + lh) * WB_STRIDE];
#pragma unroll
        for (int t = 0; t < 8; ++t) {
            v2f b = wrow[t * 16 + lm];                // one ds_load_b64
            acc[t] = __builtin_amdgcn_wmma_f32_16x16x4_f32(
                false, a, false, b, (short)0, acc[t], false, false);
        }
    }
#else
    // fallback: f16 inputs, f32 accumulate (V_WMMA_F32_16X16X32_F16)
    for (int k0 = 0; k0 < D_MODEL; k0 += 32) {
        v16h a;
#pragma unroll
        for (int j = 0; j < 16; ++j) {
            int K = j + 16 * lh;
            a[j] = (_Float16)Ar[k0 + K];
        }
#pragma unroll
        for (int t = 0; t < 8; ++t) {
            v16h b;
#pragma unroll
            for (int j = 0; j < 16; ++j) {
                int k = k0 + j + 16 * lh;
                b[j] = (_Float16)(((const float*)WB)[(((k >> 1) * WB_STRIDE) +
                                                      t * 16 + lm) * 2 + (k & 1)]);
            }
            acc[t] = __builtin_amdgcn_wmma_f32_16x16x32_f16(
                false, a, false, b, (short)0, acc[t], false, false);
        }
    }
#endif

    // C/D layout: VGPR r -> row m0 + r + 8*lh, col t*16 + lm
#pragma unroll
    for (int t = 0; t < 8; ++t) {
        float bv = bias ? bias[t * 16 + lm] : 0.0f;
#pragma unroll
        for (int r = 0; r < 8; ++r) {
            int orow = m0 + r + 8 * lh;
            if (orow < M) C[(size_t)orow * D_MODEL + t * 16 + lm] = acc[t][r] + bv;
        }
    }
}

// ---------------------------------------------------------------------------
// Fused pull-aggregation + bias + LayerNorm + ReLU + running-skip update.
// One wave per node; lane holds 4 contiguous features (float4).
// Edge loop unrolled x4 to keep 4 coalesced 512B gathers in flight.
// ---------------------------------------------------------------------------
__global__ __launch_bounds__(256) void k_agg_ln(const float* __restrict__ hin,
                                                const int* __restrict__ row_ptr,
                                                const int* __restrict__ csr_src,
                                                const float* __restrict__ dinv,
                                                const float* __restrict__ bias,
                                                const float* __restrict__ g,
                                                const float* __restrict__ bta,
                                                const float* __restrict__ base,
                                                float* __restrict__ accOut,
                                                float* __restrict__ hOut,
                                                int n) {
    int wave = threadIdx.x >> 5, lane = threadIdx.x & 31;
    int node = blockIdx.x * 8 + wave;
    if (node >= n) return;                       // wave-uniform

    const v4f* hv = (const v4f*)hin;
    float di = dinv[node];
    v4f acc = hv[(size_t)node * 32 + lane] * (di * di);   // self loop

    int e0 = row_ptr[node], e1 = row_ptr[node + 1];
    int e = e0;
    for (; e + 3 < e1; e += 4) {
        int s0 = csr_src[e + 0], s1 = csr_src[e + 1];
        int s2 = csr_src[e + 2], s3 = csr_src[e + 3];
        float w0 = di * dinv[s0], w1 = di * dinv[s1];
        float w2 = di * dinv[s2], w3 = di * dinv[s3];
        v4f h0 = hv[(size_t)s0 * 32 + lane];
        v4f h1 = hv[(size_t)s1 * 32 + lane];
        v4f h2 = hv[(size_t)s2 * 32 + lane];
        v4f h3 = hv[(size_t)s3 * 32 + lane];
        acc += h0 * w0;
        acc += h1 * w1;
        acc += h2 * w2;
        acc += h3 * w3;
    }
    for (; e < e1; ++e) {
        int s = csr_src[e];
        acc += hv[(size_t)s * 32 + lane] * (di * dinv[s]);
    }
    acc.x += bias[lane * 4 + 0];
    acc.y += bias[lane * 4 + 1];
    acc.z += bias[lane * 4 + 2];
    acc.w += bias[lane * 4 + 3];

    // LayerNorm across 128 features (wave32 butterfly reduction)
    float s1r = acc.x + acc.y + acc.z + acc.w;
    float s2r = acc.x * acc.x + acc.y * acc.y + acc.z * acc.z + acc.w * acc.w;
#pragma unroll
    for (int off = 16; off > 0; off >>= 1) {
        s1r += __shfl_xor(s1r, off, 32);
        s2r += __shfl_xor(s2r, off, 32);
    }
    float mu  = s1r * (1.0f / 128.0f);
    float var = s2r * (1.0f / 128.0f) - mu * mu;
    float inv = rsqrtf(var + 1e-5f);

    v4f out;
#pragma unroll
    for (int c = 0; c < 4; ++c) {
        float v = (acc[c] - mu) * inv * g[lane * 4 + c] + bta[lane * 4 + c];
        out[c] = v > 0.0f ? v : 0.0f;            // ReLU
    }
    if (hOut)  ((v4f*)hOut)[(size_t)node * 32 + lane] = out;
    if (accOut) {
        v4f b = ((const v4f*)base)[(size_t)node * 32 + lane];
        ((v4f*)accOut)[(size_t)node * 32 + lane] = b + out;
    }
}

// ---------------------------------------------------------------------------
// Segment mean pool: one block per graph, thread = feature dim
// ---------------------------------------------------------------------------
__global__ __launch_bounds__(128) void k_pool(const float* __restrict__ h,
                                              const int* __restrict__ ptr,
                                              float* __restrict__ g, int n_graphs) {
    int gi = blockIdx.x, d = threadIdx.x;
    if (gi >= n_graphs) return;
    int b = ptr[gi], e = ptr[gi + 1];
    float s = 0.0f;
    for (int i = b; i < e; ++i) s += h[(size_t)i * D_MODEL + d];
    g[(size_t)gi * D_MODEL + d] = s / (float)(e - b);
}

// ---------------------------------------------------------------------------
extern "C" void kernel_launch(void* const* d_in, const int* in_sizes, int n_in,
                              void* d_out, int out_size, void* d_ws, size_t ws_size,
                              hipStream_t stream) {
    const float* emb    = (const float*)d_in[0];
    const float* fc_w   = (const float*)d_in[1];
    const float* fc_b   = (const float*)d_in[2];
    const float* conv_w = (const float*)d_in[3];
    const float* conv_b = (const float*)d_in[4];
    const float* ln_g   = (const float*)d_in[5];
    const float* ln_b   = (const float*)d_in[6];
    const int*   x      = (const int*)d_in[7];
    const int*   ei     = (const int*)d_in[8];
    const int*   ptr    = (const int*)d_in[9];

    const int N  = in_sizes[7];
    const int E  = in_sizes[8] / 2;
    const int NG = in_sizes[9] - 1;
    const int* src = ei;
    const int* dst = ei + E;

    // workspace partition (256B aligned)
    auto au = [](size_t v) { return (v + 255) & ~(size_t)255; };
    char* w = (char*)d_ws;
    size_t featB = au((size_t)N * D_MODEL * sizeof(float));
    float* bufA = (float*)w; w += featB;     // feat, later running skip-sum
    float* bufB = (float*)w; w += featB;     // skip
    float* bufC = (float*)w; w += featB;     // GEMM result per layer
    float* dinv = (float*)w; w += au((size_t)N * sizeof(float));
    int* cnt     = (int*)w;  w += au((size_t)N * sizeof(int));
    int* cursor  = (int*)w;  w += au((size_t)N * sizeof(int));
    int* row_ptr = (int*)w;  w += au((size_t)(N + 1) * sizeof(int));
    int* csr_src = (int*)w;  w += au((size_t)E * sizeof(int));

    float* h_last = (float*)d_out;
    float* g_out  = (float*)d_out + (size_t)N * D_MODEL;

    const int T = 256;
    int gbN  = (N + T - 1) / T;
    int gbE  = (E + T - 1) / T;
    int gbND = (N * D_MODEL + T - 1) / T;
    int gbG  = (N + 127) / 128;   // gemm: 128 rows per block
    int gbA  = (N + 7) / 8;       // agg: 8 nodes per block

    // CSR build (reused by all 3 layers)
    k_zero2   <<<gbN, T, 0, stream>>>(cnt, cursor, N);
    k_count   <<<gbE, T, 0, stream>>>(dst, cnt, E);
    k_scan    <<<1, SCAN_T, 0, stream>>>(cnt, row_ptr, N);
    k_dinv    <<<gbN, T, 0, stream>>>(cnt, dinv, N);
    k_csr_fill<<<gbE, T, 0, stream>>>(src, dst, row_ptr, cursor, csr_src, E);

    // feat = emb[x]
    k_embed<<<gbND, T, 0, stream>>>(x, emb, bufA, N);

    // skip = feat @ fc_skip_w + fc_skip_b
    k_gemm128<<<gbG, T, 0, stream>>>(bufA, fc_w, fc_b, bufB, N);

    // layer 0: tmp = feat @ W0 ; acc = skip + relu(LN(agg(tmp)+b0))
    k_gemm128<<<gbG, T, 0, stream>>>(bufA, conv_w, nullptr, bufC, N);
    k_agg_ln <<<gbA, T, 0, stream>>>(bufC, row_ptr, csr_src, dinv,
                                     conv_b, ln_g, ln_b,
                                     bufB, bufA, nullptr, N);

    // layer 1: tmp = acc @ W1 ; acc += relu(LN(agg(tmp)+b1))
    k_gemm128<<<gbG, T, 0, stream>>>(bufA, conv_w + 1 * D_MODEL * D_MODEL, nullptr, bufC, N);
    k_agg_ln <<<gbA, T, 0, stream>>>(bufC, row_ptr, csr_src, dinv,
                                     conv_b + D_MODEL, ln_g + D_MODEL, ln_b + D_MODEL,
                                     bufA, bufA, nullptr, N);

    // layer 2: h_last = relu(LN(agg(acc @ W2)+b2))
    k_gemm128<<<gbG, T, 0, stream>>>(bufA, conv_w + 2 * D_MODEL * D_MODEL, nullptr, bufC, N);
    k_agg_ln <<<gbA, T, 0, stream>>>(bufC, row_ptr, csr_src, dinv,
                                     conv_b + 2 * D_MODEL, ln_g + 2 * D_MODEL, ln_b + 2 * D_MODEL,
                                     nullptr, nullptr, h_last, N);

    // avg pool per graph
    k_pool<<<NG, 128, 0, stream>>>(h_last, ptr, g_out, NG);
}
